// PEER_90340342104129
// MI455X (gfx1250) — matrix operand order repaired and checked by reading, not compiled, not run on hardware
//
#include <hip/hip_runtime.h>
#include <hip/hip_bf16.h>
#include <math.h>

// ---------------------------------------------------------------------------
// PEER layer for MI455X (gfx1250, wave32, WMMA).
// Pipeline: rmsnorm -> bf16 convert -> WMMA q-GEMM -> WMMA sim-GEMMs ->
//           wave32 top-16 -> product-key combine+softmax -> expert gather/apply
// ---------------------------------------------------------------------------

typedef __attribute__((ext_vector_type(16))) __bf16 v16bf;
typedef __attribute__((ext_vector_type(8)))  __bf16 v8bf;
typedef __attribute__((ext_vector_type(8)))  float  v8f;

#define T_TOK 1024   // B*N = 2*512
#define DIM   1024
#define HEADS 8
#define NKEYS 256
#define DKEY  512
#define TOPK  16
#define QCOLS 8192   // 2*H*DKEY
#define PCOLS 4096   // H*DKEY
#define NEG_INF (-3.0e38f)

// -------------------------- RMS norm ---------------------------------------
__global__ __launch_bounds__(256) void k_rmsnorm(const float* __restrict__ x,
                                                 const float* __restrict__ gamma,
                                                 float* __restrict__ xn,
                                                 __bf16* __restrict__ xnb) {
    int t = blockIdx.x, tid = threadIdx.x;
    int lane = tid & 31, w = tid >> 5;
    const float4* xr = (const float4*)(x + (size_t)t * DIM);
    float4 v = xr[tid];
    float ss = v.x * v.x + v.y * v.y + v.z * v.z + v.w * v.w;
#pragma unroll
    for (int o = 16; o; o >>= 1) ss += __shfl_xor(ss, o, 32);
    __shared__ float red[8];
    if (lane == 0) red[w] = ss;
    __syncthreads();
    float tot = red[0] + red[1] + red[2] + red[3] + red[4] + red[5] + red[6] + red[7];
    // v / ||v|| * gamma * sqrt(D);  sqrt(1024) = 32
    float scale = 32.0f / fmaxf(sqrtf(tot), 1e-12f);
    int d = tid * 4;
    float y0 = v.x * scale * gamma[d + 0];
    float y1 = v.y * scale * gamma[d + 1];
    float y2 = v.z * scale * gamma[d + 2];
    float y3 = v.w * scale * gamma[d + 3];
    float4* xo = (float4*)(xn + (size_t)t * DIM);
    xo[tid] = make_float4(y0, y1, y2, y3);
    __bf16* bo = xnb + (size_t)t * DIM + d;
    bo[0] = (__bf16)y0; bo[1] = (__bf16)y1; bo[2] = (__bf16)y2; bo[3] = (__bf16)y3;
}

// -------------------------- f32 -> bf16 convert ----------------------------
__global__ __launch_bounds__(256) void k_cvt_bf16(const float* __restrict__ in,
                                                  __bf16* __restrict__ out, int n) {
    int i = (blockIdx.x * 256 + threadIdx.x) * 4;
    if (i < n) {
        float4 v = *(const float4*)(in + i);
        out[i + 0] = (__bf16)v.x;
        out[i + 1] = (__bf16)v.y;
        out[i + 2] = (__bf16)v.z;
        out[i + 3] = (__bf16)v.w;
    }
}

// ---- A-fragment: 16x32 bf16, lane<16 -> K {0..15}, lane>=16 -> +16 ----
__device__ __forceinline__ v16bf make_a_frag(const __bf16* ap /* lane-adjusted */,
                                             int k0) {
    v8bf lo = *(const v8bf*)(ap + k0);
    v8bf hi = *(const v8bf*)(ap + k0 + 8);
    v16bf a;
#pragma unroll
    for (int i = 0; i < 8; i++) { a[i] = lo[i]; a[8 + i] = hi[i]; }
    return a;
}

// -------------------------- Q GEMM: xn[1024x1024] @ wq^T[1024x8192] --------
// Output directly remapped+converted to bf16 q[2][T][4096] for the sim GEMM.
__global__ __launch_bounds__(256) void k_gemm_q(const __bf16* __restrict__ A,
                                                const __bf16* __restrict__ Wq,
                                                __bf16* __restrict__ Q) {
    int lane = threadIdx.x & 31, w = threadIdx.x >> 5;
    int m0 = blockIdx.y * 16;
    int n0 = blockIdx.x * 128 + w * 16;
    // A: row m0+(lane&15), K-halves selected by lane>=16
    const __bf16* ap = A + (size_t)(m0 + (lane & 15)) * DIM + ((lane >> 4) << 4);
    // B col n = wq row n (contraction over its contiguous K axis)
    const __bf16* bp = Wq + (size_t)(n0 + (lane & 15)) * DIM + ((lane >> 4) << 4);
    v8f acc = {};
    for (int k0 = 0; k0 < DIM; k0 += 32) {
        v16bf a = make_a_frag(ap, k0);
        v16bf b = *(const v16bf*)(bp + k0);
        acc = __builtin_amdgcn_wmma_f32_16x16x32_bf16(false, a, false, b,
                                                      (short)0, acc, false, false);
    }
    int mrow = m0 + ((lane >> 4) << 3);
    int n = n0 + (lane & 15);
    int p = n >> 12, rem = n & 4095;           // col = p*4096 + h*512 + dk
    __bf16* qb = Q + (size_t)p * T_TOK * PCOLS + rem;
#pragma unroll
    for (int r = 0; r < 8; r++)
        qb[(size_t)(mrow + r) * PCOLS] = (__bf16)acc[r];
}

// -------------------------- sim GEMMs: q[p,:,h,:] @ keys[h,:,p,:]^T --------
__global__ __launch_bounds__(256) void k_gemm_sim(const __bf16* __restrict__ Q,
                                                  const __bf16* __restrict__ Kb,
                                                  float* __restrict__ S) {
    int lane = threadIdx.x & 31, w = threadIdx.x >> 5;
    int hp = blockIdx.z, h = hp >> 1, p = hp & 1;
    int m0 = blockIdx.y * 16;
    int n0 = blockIdx.x * 128 + w * 16;
    const __bf16* ap = Q + (size_t)p * T_TOK * PCOLS + (size_t)h * DKEY
                         + (size_t)(m0 + (lane & 15)) * PCOLS + ((lane >> 4) << 4);
    const __bf16* bp = Kb + ((size_t)(h * NKEYS + n0 + (lane & 15)) * 2 + p) * DKEY
                          + ((lane >> 4) << 4);
    v8f acc = {};
    for (int k0 = 0; k0 < DKEY; k0 += 32) {
        v16bf a = make_a_frag(ap, k0);
        v16bf b = *(const v16bf*)(bp + k0);
        acc = __builtin_amdgcn_wmma_f32_16x16x32_bf16(false, a, false, b,
                                                      (short)0, acc, false, false);
    }
    int mrow = m0 + ((lane >> 4) << 3);
    int n = n0 + (lane & 15);
#pragma unroll
    for (int r = 0; r < 8; r++)
        S[(((size_t)p * T_TOK + mrow + r) * HEADS + h) * NKEYS + n] = acc[r];
}

// ----------------- top-16 of 256 per (p,t,h): one wave32 per row -----------
__global__ __launch_bounds__(256) void k_topk256(const float* __restrict__ S,
                                                 float* __restrict__ sv,
                                                 int* __restrict__ si) {
    int gw = (blockIdx.x * 256 + threadIdx.x) >> 5;   // row id, 16384 rows
    int lane = threadIdx.x & 31;
    const float* row = S + (size_t)gw * NKEYS;
    float v[8];
#pragma unroll
    for (int i = 0; i < 8; i++) v[i] = row[lane + 32 * i];
    float* svr = sv + (size_t)gw * TOPK;
    int*   sir = si + (size_t)gw * TOPK;
    for (int t = 0; t < TOPK; t++) {
        float bm = NEG_INF; int bs = 0;
#pragma unroll
        for (int i = 0; i < 8; i++) if (v[i] > bm) { bm = v[i]; bs = i; }
        int bi = lane + 32 * bs;
        float m = bm; int mi = bi;
#pragma unroll
        for (int o = 16; o; o >>= 1) {               // lowest-index tie-break
            float om = __shfl_xor(m, o, 32);
            int   oi = __shfl_xor(mi, o, 32);
            if (om > m || (om == m && oi < mi)) { m = om; mi = oi; }
        }
        if (lane == 0) { svr[t] = m; sir[t] = mi; }
        if (mi == bi) v[bs] = NEG_INF;
    }
}

// ------- combine product keys: outer sum, top-16 of 256, softmax -----------
__global__ __launch_bounds__(256) void k_combine(const float* __restrict__ sv,
                                                 const int* __restrict__ si,
                                                 float* __restrict__ ew,
                                                 int* __restrict__ ei) {
    int gw = (blockIdx.x * 256 + threadIdx.x) >> 5;   // (t*8 + h), 8192 rows
    int lane = threadIdx.x & 31;
    int t = gw >> 3, h = gw & 7;
    size_t rx = ((size_t)0 * T_TOK + t) * HEADS + h;
    size_t ry = ((size_t)1 * T_TOK + t) * HEADS + h;
    float sx_l = 0.f, sy_l = 0.f; int ix_l = 0, iy_l = 0;
    if (lane < 16) {
        sx_l = sv[rx * TOPK + lane]; ix_l = si[rx * TOPK + lane];
        sy_l = sv[ry * TOPK + lane]; iy_l = si[ry * TOPK + lane];
    }
    float cs[8];
#pragma unroll
    for (int u = 0; u < 8; u++) {                     // candidate c = a*16 + b
        int c = lane + 32 * u, a = c >> 4, b = c & 15;
        cs[u] = __shfl(sx_l, a, 32) + __shfl(sy_l, b, 32);
    }
    float myscore = 0.f, smax = 0.f, sum = 0.f;
    int myeid = 0;
    for (int ts = 0; ts < TOPK; ts++) {
        float bm = NEG_INF; int bs = 0;
#pragma unroll
        for (int u = 0; u < 8; u++) if (cs[u] > bm) { bm = cs[u]; bs = u; }
        int bi = lane + 32 * bs;
        float m = bm; int mi = bi;
#pragma unroll
        for (int o = 16; o; o >>= 1) {
            float om = __shfl_xor(m, o, 32);
            int   oi = __shfl_xor(mi, o, 32);
            if (om > m || (om == m && oi < mi)) { m = om; mi = oi; }
        }
        int a = mi >> 4, b = mi & 15;
        int eid = __shfl(ix_l, a, 32) * NKEYS + __shfl(iy_l, b, 32);
        if (ts == 0) smax = m;                        // descending -> first is max
        sum += expf(m - smax);
        if (lane == ts) { myscore = m; myeid = eid; }
        if (mi == bi) cs[bs] = NEG_INF;
    }
    if (lane < TOPK) {
        size_t o = (size_t)gw * TOPK + lane;
        ew[o] = expf(myscore - smax) / sum;
        ei[o] = myeid;
    }
}

// --------- expert gather/apply: block per token (HBM-gather bound) ---------
__global__ __launch_bounds__(256) void k_expert(const float* __restrict__ xn,
                                                const float* __restrict__ tdown,
                                                const float* __restrict__ tup,
                                                const int* __restrict__ ei,
                                                const float* __restrict__ ew,
                                                float* __restrict__ out) {
    __shared__ float xs[DIM];
    __shared__ float coeff[HEADS * TOPK];
    __shared__ int   eids[HEADS * TOPK];
    int t = blockIdx.x, tid = threadIdx.x;
    int lane = tid & 31, h = tid >> 5;               // wave == head
    ((float4*)xs)[tid] = ((const float4*)(xn + (size_t)t * DIM))[tid];
    if (tid < HEADS * TOPK) eids[tid] = ei[(size_t)t * HEADS * TOPK + tid];
    __syncthreads();

    const float4* xs4 = (const float4*)xs;
    for (int k = 0; k < TOPK; k++) {
        int j = h * TOPK + k;
        int e = eids[j];
        if (k + 1 < TOPK)                            // stream the gather
            __builtin_prefetch(tdown + (size_t)eids[j + 1] * DIM + lane * 32, 0, 0);
        const float4* wd = (const float4*)(tdown + (size_t)e * DIM);
        float s = 0.f;
#pragma unroll
        for (int i = 0; i < 8; i++) {
            float4 wv = wd[lane + 32 * i];
            float4 xv = xs4[lane + 32 * i];
            s += wv.x * xv.x + wv.y * xv.y + wv.z * xv.z + wv.w * xv.w;
        }
#pragma unroll
        for (int o = 16; o; o >>= 1) s += __shfl_xor(s, o, 32);
        if (lane == 0) {
            float g = 0.5f * s * (1.0f + erff(s * 0.70710678118654752f)); // exact gelu
            coeff[j] = g * ew[(size_t)t * HEADS * TOPK + j];
        }
    }
    __syncthreads();

    float4 acc = make_float4(0.f, 0.f, 0.f, 0.f);
    const float4* up = (const float4*)tup;
    for (int j = 0; j < HEADS * TOPK; j++) {
        if (j + 4 < HEADS * TOPK)
            __builtin_prefetch(&up[(size_t)eids[j + 4] * (DIM / 4) + tid], 0, 0);
        float4 v = up[(size_t)eids[j] * (DIM / 4) + tid];
        float c = coeff[j];
        acc.x += c * v.x; acc.y += c * v.y; acc.z += c * v.z; acc.w += c * v.w;
    }
    ((float4*)out)[(size_t)t * (DIM / 4) + tid] = acc;
}

// ---------------------------------------------------------------------------
extern "C" void kernel_launch(void* const* d_in, const int* in_sizes, int n_in,
                              void* d_out, int out_size, void* d_ws, size_t ws_size,
                              hipStream_t stream) {
    const float* x      = (const float*)d_in[0];
    const float* gamma  = (const float*)d_in[1];
    const float* wq     = (const float*)d_in[2];
    const float* keys   = (const float*)d_in[3];
    const float* tdown  = (const float*)d_in[4];
    const float* tup    = (const float*)d_in[5];
    float* out = (float*)d_out;

    char* ws = (char*)d_ws;
    size_t off = 0;
    auto alloc = [&](size_t bytes) { char* p = ws + off; off = (off + bytes + 255) & ~(size_t)255; return p; };
    float*  xn_f  = (float*) alloc((size_t)T_TOK * DIM * 4);
    __bf16* xn_b  = (__bf16*)alloc((size_t)T_TOK * DIM * 2);
    __bf16* wq_b  = (__bf16*)alloc((size_t)QCOLS * DIM * 2);
    __bf16* keys_b= (__bf16*)alloc((size_t)HEADS * NKEYS * 2 * DKEY * 2);
    __bf16* q_b   = (__bf16*)alloc((size_t)2 * T_TOK * PCOLS * 2);
    float*  sim   = (float*) alloc((size_t)2 * T_TOK * HEADS * NKEYS * 4);
    float*  sv    = (float*) alloc((size_t)2 * T_TOK * HEADS * TOPK * 4);
    int*    si    = (int*)   alloc((size_t)2 * T_TOK * HEADS * TOPK * 4);
    float*  ew    = (float*) alloc((size_t)T_TOK * HEADS * TOPK * 4);
    int*    ei    = (int*)   alloc((size_t)T_TOK * HEADS * TOPK * 4);

    k_rmsnorm<<<T_TOK, 256, 0, stream>>>(x, gamma, xn_f, xn_b);

    int n_wq = QCOLS * DIM;                 // 8,388,608
    k_cvt_bf16<<<n_wq / 1024, 256, 0, stream>>>(wq, wq_b, n_wq);
    int n_keys = HEADS * NKEYS * 2 * DKEY;  // 2,097,152
    k_cvt_bf16<<<n_keys / 1024, 256, 0, stream>>>(keys, keys_b, n_keys);

    k_gemm_q<<<dim3(QCOLS / 128, T_TOK / 16), 256, 0, stream>>>(xn_b, wq_b, q_b);
    k_gemm_sim<<<dim3(NKEYS / 128, T_TOK / 16, HEADS * 2), 256, 0, stream>>>(q_b, keys_b, sim);

    k_topk256<<<(2 * T_TOK * HEADS) / 8, 256, 0, stream>>>(sim, sv, si);
    k_combine<<<(T_TOK * HEADS) / 8, 256, 0, stream>>>(sv, si, ew, ei);

    k_expert<<<T_TOK, 256, 0, stream>>>(xn_f, tdown, tup, ei, ew, out);
}
